// SS2D_16939351015734
// MI455X (gfx1250) — compile-verified, matrix operand order checked by reading, not run on hardware
//
#include <hip/hip_runtime.h>
#include <hip/hip_bf16.h>
#include <math.h>

// ---------------------------------------------------------------------------
// SS2D (Mamba-style selective-scan 2D block) for gfx1250 / MI455X.
// Dense projections use v_wmma_f32_16x16x32_bf16 (wave32 WMMA, NT tiles/wave,
// float4 fragment loads); the sequential selective scan is f32 VALU with LDS
// staging of the shared B/C rows.
// ---------------------------------------------------------------------------

#define BATCH   4
#define HH      64
#define WW      64
#define LSEQ    4096                  // HH*WW
#define MROWS   16384                 // BATCH*LSEQ
#define DMODEL  192
#define DINNER  384
#define NSTATE  16
#define DTRANK  12
#define DBL     44                    // DTRANK + 2*NSTATE
#define DBLP    48                    // padded to 16
#define GHID    96

typedef __attribute__((ext_vector_type(16))) __bf16 v16bf;
typedef __attribute__((ext_vector_type(8)))  float  v8f;

// Pack 8 consecutive-k elements (two aligned float4 chunks) into fragment
// elements [base, base+8).
__device__ __forceinline__ void pack8(v16bf& f, int base, const float4 lo,
                                      const float4 hi)
{
    f[base + 0] = (__bf16)lo.x;
    f[base + 1] = (__bf16)lo.y;
    f[base + 2] = (__bf16)lo.z;
    f[base + 3] = (__bf16)lo.w;
    f[base + 4] = (__bf16)hi.x;
    f[base + 5] = (__bf16)hi.y;
    f[base + 6] = (__bf16)hi.z;
    f[base + 7] = (__bf16)hi.w;
}

// Scaled variant (scale = 0/1 implements N-edge zero padding branch-free).
__device__ __forceinline__ void pack8s(v16bf& f, int base, const float4 lo,
                                       const float4 hi, const float s)
{
    f[base + 0] = (__bf16)(lo.x * s);
    f[base + 1] = (__bf16)(lo.y * s);
    f[base + 2] = (__bf16)(lo.z * s);
    f[base + 3] = (__bf16)(lo.w * s);
    f[base + 4] = (__bf16)(hi.x * s);
    f[base + 5] = (__bf16)(hi.y * s);
    f[base + 6] = (__bf16)(hi.z * s);
    f[base + 7] = (__bf16)(hi.w * s);
}

// ---------------------------------------------------------------------------
// C = A(MxK) * W(NxK)^T.  One wave computes a 16x(16*NT) strip of C:
// the A fragment is loaded once per 32-wide k-step and reused by NT WMMAs.
// K % 32 == 0, lda == ldw == K, rows of A/W 32B-aligned (K % 8 == 0).
// 16-bit A 16x32 fragment layout (ISA 7.12.2): lane&15 = row/col,
// lane>>4 = k-half, element i -> k = ((i&8)<<1) + 8*(lane>>4) + (i&7),
// i.e. two contiguous 8-float chunks at k0 and k0+16.
// PAD=false: all NT*16 output columns valid (no clamp/scale/predication).
// ---------------------------------------------------------------------------
template <int NT, bool PAD>
__global__ __launch_bounds__(32)
void wmma_gemm_nt(const float* __restrict__ A, const float* __restrict__ W,
                  float* __restrict__ C, int K, int ngroups, int Nreal, int ldc)
{
    const int tile = blockIdx.x;
    const int mt   = tile / ngroups;
    const int ng   = tile - mt * ngroups;
    const int m0   = mt << 4;
    const int n0   = ng * (16 * NT);
    const int lane = threadIdx.x;
    const int r     = lane & 15;
    const int khalf = lane >> 4;

    const float* __restrict__ Arow = A + (size_t)(m0 + r) * K;

    const float* Wrow[NT];
    float wscale[NT];
#pragma unroll
    for (int s = 0; s < NT; ++s) {
        const int n = n0 + s * 16 + r;
        if (PAD) {
            const bool v = n < Nreal;
            Wrow[s]   = W + (size_t)(v ? n : 0) * K;   // clamp: always in-bounds
            wscale[s] = v ? 1.0f : 0.0f;               // zero pad, branch-free
        } else {
            Wrow[s]   = W + (size_t)n * K;
            wscale[s] = 1.0f;
        }
    }

    v8f c[NT];
#pragma unroll
    for (int s = 0; s < NT; ++s) c[s] = (v8f){};

    for (int kk = 0; kk < K; kk += 32) {
        const int k0 = kk + (khalf << 3);
        // A fragment: 4 x global_load_b128
        const float4* a0 = (const float4*)(Arow + k0);
        const float4* a1 = (const float4*)(Arow + k0 + 16);
        v16bf a;
        pack8(a, 0, a0[0], a0[1]);
        pack8(a, 8, a1[0], a1[1]);
#pragma unroll
        for (int s = 0; s < NT; ++s) {
            const float4* b0 = (const float4*)(Wrow[s] + k0);
            const float4* b1 = (const float4*)(Wrow[s] + k0 + 16);
            v16bf b;
            if (PAD) {
                pack8s(b, 0, b0[0], b0[1], wscale[s]);
                pack8s(b, 8, b1[0], b1[1], wscale[s]);
            } else {
                pack8(b, 0, b0[0], b0[1]);
                pack8(b, 8, b1[0], b1[1]);
            }
            c[s] = __builtin_amdgcn_wmma_f32_16x16x32_bf16(
                       false, a, false, b, (short)0, c[s], false, false);
        }
    }

    // C/D layout: VGPR j, lane l -> (m = j + 8*(l>>4), n = l&15)
#pragma unroll
    for (int s = 0; s < NT; ++s) {
        const int n = n0 + s * 16 + r;
        if (!PAD || n < Nreal) {
#pragma unroll
            for (int j = 0; j < 8; ++j) {
                const int m = m0 + j + (khalf << 3);
                C[(size_t)m * ldc + n] = c[s][j];
            }
        }
    }
}

// ---------------------------------------------------------------------------
// Depthwise 3x3 conv (same padding) + bias + SiLU over the xi half of xz.
// xz: (M,768) row-major; output xs: (M,384) pixel-major.
// ---------------------------------------------------------------------------
__global__ __launch_bounds__(256)
void dwconv2d_silu_k(const float* __restrict__ xz, const float* __restrict__ cw,
                     const float* __restrict__ cb, float* __restrict__ xs)
{
    const int t = blockIdx.x * 256 + threadIdx.x;
    if (t >= MROWS * DINNER) return;
    const int d   = t % DINNER;
    const int pix = t / DINNER;
    const int wx  = pix % WW;
    const int hy  = (pix / WW) % HH;
    const int b   = pix / LSEQ;

    float acc = cb[d];
#pragma unroll
    for (int dy = 0; dy < 3; ++dy) {
        const int h2 = hy + dy - 1;
        if (h2 < 0 || h2 >= HH) continue;
#pragma unroll
        for (int dx = 0; dx < 3; ++dx) {
            const int w2 = wx + dx - 1;
            if (w2 < 0 || w2 >= WW) continue;
            acc += xz[(size_t)((b * HH + h2) * WW + w2) * (2 * DINNER) + d]
                 * cw[d * 9 + dy * 3 + dx];
        }
    }
    xs[(size_t)pix * DINNER + d] = acc / (1.0f + __expf(-acc));   // SiLU
}

// ---------------------------------------------------------------------------
// SimpleGate1D for Bs_low / Cs_low (16 -> 2*96 -> gelu-gate -> 16),
// result added in place into the B/C channels of x_dbl.
// ---------------------------------------------------------------------------
__global__ __launch_bounds__(256)
void gates_add_k(const float* __restrict__ low_dbl,
                 const float* __restrict__ gb1, const float* __restrict__ gb2,
                 const float* __restrict__ gc1, const float* __restrict__ gc2,
                 float* __restrict__ x_dbl)
{
    __shared__ float s1b[2 * GHID * NSTATE];   // (192,16)
    __shared__ float s2b[NSTATE * GHID];       // (16,96)
    __shared__ float s1c[2 * GHID * NSTATE];
    __shared__ float s2c[NSTATE * GHID];
    const int tid = threadIdx.x;
    for (int i = tid; i < 2 * GHID * NSTATE; i += 256) { s1b[i] = gb1[i]; s1c[i] = gc1[i]; }
    for (int i = tid; i < NSTATE * GHID; i += 256)     { s2b[i] = gb2[i]; s2c[i] = gc2[i]; }
    __syncthreads();

    const int m = blockIdx.x * 256 + tid;      // M is a multiple of 256
    const float* __restrict__ row = low_dbl + (size_t)m * DBLP;
    float vB[NSTATE], vC[NSTATE], aB[NSTATE], aC[NSTATE];
#pragma unroll
    for (int n = 0; n < NSTATE; ++n) {
        vB[n] = row[DTRANK + n];
        vC[n] = row[DTRANK + NSTATE + n];
        aB[n] = 0.0f; aC[n] = 0.0f;
    }
    for (int j = 0; j < GHID; ++j) {
        float h1b = 0.f, h2b = 0.f, h1c = 0.f, h2c = 0.f;
#pragma unroll
        for (int n = 0; n < NSTATE; ++n) {
            h1b += s1b[j * NSTATE + n] * vB[n];
            h2b += s1b[(GHID + j) * NSTATE + n] * vB[n];
            h1c += s1c[j * NSTATE + n] * vC[n];
            h2c += s1c[(GHID + j) * NSTATE + n] * vC[n];
        }
        const float gb = 0.5f * h1b * (1.0f + erff(h1b * 0.70710678118654752f)) * h2b;
        const float gc = 0.5f * h1c * (1.0f + erff(h1c * 0.70710678118654752f)) * h2c;
#pragma unroll
        for (int o = 0; o < NSTATE; ++o) {
            aB[o] += s2b[o * GHID + j] * gb;
            aC[o] += s2c[o * GHID + j] * gc;
        }
    }
    float* __restrict__ orow = x_dbl + (size_t)m * DBLP;
#pragma unroll
    for (int o = 0; o < NSTATE; ++o) {
        orow[DTRANK + o]          += aB[o];
        orow[DTRANK + NSTATE + o] += aC[o];
    }
}

// ---------------------------------------------------------------------------
// Dilated depthwise conv1d over L (k=15, dilation=2, pad=14 -> "same").
// ---------------------------------------------------------------------------
__global__ __launch_bounds__(256)
void conv1d_k(const float* __restrict__ x_dbl, const float* __restrict__ wdt,
              const float* __restrict__ wB, const float* __restrict__ wC,
              float* __restrict__ outc)
{
    const int t = blockIdx.x * 256 + threadIdx.x;
    if (t >= MROWS * DBL) return;
    const int c = t % DBL;
    const int m = t / DBL;
    const int b = m >> 12;             // / LSEQ
    const int l = m & (LSEQ - 1);
    const float* __restrict__ wr =
        (c < DTRANK) ? (wdt + c * 15)
      : (c < DTRANK + NSTATE) ? (wB + (c - DTRANK) * 15)
      : (wC + (c - DTRANK - NSTATE) * 15);
    const float* __restrict__ base = x_dbl + ((size_t)b << 12) * DBLP + c;
    float acc = 0.0f;
#pragma unroll
    for (int i = 0; i < 15; ++i) {
        const int li = l + 2 * (i - 7);
        if (li >= 0 && li < LSEQ) acc += base[(size_t)li * DBLP] * wr[i];
    }
    outc[(size_t)m * DBLP + c] = acc;
}

// ---------------------------------------------------------------------------
// delta = softplus(dt_proj_w @ dts_conv + dt_proj_b)    (K = 12, plain VALU)
// ---------------------------------------------------------------------------
__global__ __launch_bounds__(256)
void delta_k(const float* __restrict__ convo, const float* __restrict__ dtw,
             const float* __restrict__ dtb, float* __restrict__ delta)
{
    const int t = blockIdx.x * 256 + threadIdx.x;
    if (t >= MROWS * DINNER) return;
    const int d = t % DINNER;
    const int m = t / DINNER;
    float acc = dtb[d];
    const float* __restrict__ r = convo + (size_t)m * DBLP;
#pragma unroll
    for (int k = 0; k < DTRANK; ++k) acc += dtw[d * DTRANK + k] * r[k];
    delta[t] = (acc > 20.0f) ? acc : log1pf(__expf(acc));
}

// ---------------------------------------------------------------------------
// Selective scan: one block per batch, one thread per inner channel d.
// B/C rows for a 64-step chunk are staged in LDS (shared across all d).
// ---------------------------------------------------------------------------
#define SCAN_CH 64
__global__ __launch_bounds__(DINNER)
void scan_k(const float* __restrict__ delta, const float* __restrict__ xs,
            const float* __restrict__ convo, const float* __restrict__ A_logs,
            const float* __restrict__ Ds, float* __restrict__ y)
{
    __shared__ float sB[SCAN_CH * NSTATE];
    __shared__ float sC[SCAN_CH * NSTATE];
    const int b = blockIdx.x;
    const int d = threadIdx.x;

    float Arow[NSTATE], h[NSTATE];
#pragma unroll
    for (int n = 0; n < NSTATE; ++n) {
        Arow[n] = -__expf(A_logs[d * NSTATE + n]);   // A = -exp(A_logs)
        h[n] = 0.0f;
    }
    const float Dd = Ds[d];

    for (int l0 = 0; l0 < LSEQ; l0 += SCAN_CH) {
        __syncthreads();
        for (int i = d; i < SCAN_CH * NSTATE; i += DINNER) {
            const int li = i >> 4, n = i & 15;
            const size_t mrow = ((size_t)b * LSEQ + l0 + li) * DBLP;
            sB[i] = convo[mrow + DTRANK + n];
            sC[i] = convo[mrow + DTRANK + NSTATE + n];
        }
        if (l0 + SCAN_CH < LSEQ) {   // pre-pull next chunk (global_prefetch_b8)
            const size_t nrow = ((size_t)b * LSEQ + l0 + SCAN_CH + (d >> 4)) * DBLP;
            __builtin_prefetch(convo + nrow, 0, 1);
        }
        __syncthreads();
        for (int li = 0; li < SCAN_CH; ++li) {
            const size_t m  = (size_t)b * LSEQ + l0 + li;
            const float  dl = delta[m * DINNER + d];
            const float  u  = xs[m * DINNER + d];
            const float  du = dl * u;
            float yy = 0.0f;
#pragma unroll
            for (int n = 0; n < NSTATE; ++n) {
                h[n] = __expf(dl * Arow[n]) * h[n] + du * sB[li * NSTATE + n];
                yy  += h[n] * sC[li * NSTATE + n];
            }
            y[m * DINNER + d] = yy + u * Dd;
        }
    }
}

// ---------------------------------------------------------------------------
// y *= silu(z)   (z = second half of xz), in place.
// ---------------------------------------------------------------------------
__global__ __launch_bounds__(256)
void gate_z_k(const float* __restrict__ xz, float* __restrict__ y)
{
    const int t = blockIdx.x * 256 + threadIdx.x;
    if (t >= MROWS * DINNER) return;
    const int d = t % DINNER;
    const int m = t / DINNER;
    const float z = xz[(size_t)m * (2 * DINNER) + DINNER + d];
    y[t] *= z / (1.0f + __expf(-z));
}

// ---------------------------------------------------------------------------
extern "C" void kernel_launch(void* const* d_in, const int* in_sizes, int n_in,
                              void* d_out, int out_size, void* d_ws, size_t ws_size,
                              hipStream_t stream)
{
    const float* x         = (const float*)d_in[0];
    const float* hbl       = (const float*)d_in[1];
    const float* W_in      = (const float*)d_in[2];
    const float* W_in_low  = (const float*)d_in[3];
    const float* conv2d_w  = (const float*)d_in[4];
    const float* conv2d_b  = (const float*)d_in[5];
    const float* x_proj_w  = (const float*)d_in[6];
    const float* x_proj_wl = (const float*)d_in[7];
    const float* conv_dt_w = (const float*)d_in[8];
    const float* conv_B_w  = (const float*)d_in[9];
    const float* conv_C_w  = (const float*)d_in[10];
    const float* gb_fc1    = (const float*)d_in[11];
    const float* gb_fc2    = (const float*)d_in[12];
    const float* gc_fc1    = (const float*)d_in[13];
    const float* gc_fc2    = (const float*)d_in[14];
    const float* dt_proj_w = (const float*)d_in[15];
    const float* dt_proj_b = (const float*)d_in[16];
    const float* A_logs    = (const float*)d_in[17];
    const float* Ds        = (const float*)d_in[18];
    const float* W_out     = (const float*)d_in[19];
    float* out = (float*)d_out;

    // Workspace layout (floats).
    float* ws      = (float*)d_ws;
    const size_t M = MROWS;
    float* xz      = ws;                       // M*768
    float* low     = xz + M * 768;             // M*384  (reused as delta)
    float* xs      = low + M * 384;            // M*384
    float* x_dbl   = xs + M * 384;             // M*48
    float* low_dbl = x_dbl + M * DBLP;         // M*48
    float* convo   = low_dbl + M * DBLP;       // M*48
    float* y       = convo + M * DBLP;         // M*384
    float* delta   = low;                      // alias: low consumed before K6

    const int MT = MROWS / 16;                 // 1024 m-tiles

    // 1) in_proj: xz = x @ W_in^T  (K=192, N=768, 4 tiles/wave, no padding)
    wmma_gemm_nt<4, false><<<MT * (768 / 64), 32, 0, stream>>>(x, W_in, xz, DMODEL, 768 / 64, 768, 768);
    // 1b) low = hbl @ W_in_low^T  (K=192, N=384)
    wmma_gemm_nt<4, false><<<MT * (384 / 64), 32, 0, stream>>>(hbl, W_in_low, low, DMODEL, 384 / 64, 384, 384);
    // 2) depthwise conv2d + SiLU -> xs
    dwconv2d_silu_k<<<(MROWS * DINNER) / 256, 256, 0, stream>>>(xz, conv2d_w, conv2d_b, xs);
    // 3a) x_dbl = xs @ x_proj_w^T      (K=384, N=44 padded to 48, 3 tiles/wave)
    wmma_gemm_nt<3, true><<<MT, 32, 0, stream>>>(xs, x_proj_w, x_dbl, DINNER, 1, DBL, DBLP);
    // 3b) low_dbl = low @ x_proj_w_low^T
    wmma_gemm_nt<3, true><<<MT, 32, 0, stream>>>(low, x_proj_wl, low_dbl, DINNER, 1, DBL, DBLP);
    // 4) SimpleGate(Bs_low), SimpleGate(Cs_low) added into x_dbl
    gates_add_k<<<MROWS / 256, 256, 0, stream>>>(low_dbl, gb_fc1, gb_fc2, gc_fc1, gc_fc2, x_dbl);
    // 5) dilated depthwise conv1d over L for dt/B/C
    conv1d_k<<<(MROWS * DBL + 255) / 256, 256, 0, stream>>>(x_dbl, conv_dt_w, conv_B_w, conv_C_w, convo);
    // 6) delta = softplus(dt_proj(dts) + b)
    delta_k<<<(MROWS * DINNER) / 256, 256, 0, stream>>>(convo, dt_proj_w, dt_proj_b, delta);
    // 7) selective scan -> y (+ u*D)
    scan_k<<<BATCH, DINNER, 0, stream>>>(delta, xs, convo, A_logs, Ds, y);
    // 8a) y *= silu(z)
    gate_z_k<<<(MROWS * DINNER) / 256, 256, 0, stream>>>(xz, y);
    // 8b) out = y @ W_out^T  (K=384, N=192, 4 tiles/wave)
    wmma_gemm_nt<4, false><<<MT * (DMODEL / 64), 32, 0, stream>>>(y, W_out, out, DINNER, DMODEL / 64, DMODEL, DMODEL);
}
